// SelfAttention_79731772883764
// MI455X (gfx1250) — matrix operand order repaired and checked by reading, not compiled, not run on hardware
//
#include <hip/hip_runtime.h>
#include <hip/hip_bf16.h>

// ---------------------------------------------------------------------------
// Self-attention (N=4, C=256, H=W=64 -> HW=4096, OC=32) for gfx1250 (MI455X)
// Pipeline:
//   k0a: Wv fp32 -> bf16
//   k0b: x  fp32 [n][c][p] -> xT bf16 [n][p][c]   (B-operand layout for V proj)
//   k1 : Q/K = Wq/Wk * x + b  (fp32 VALU, tiny)   -> bf16 [n][p][d]
//   k2 : V = Wv * X + bv via WMMA bf16            -> bf16 [n][c][p]
//   k3 : fused attention per 16-row query strip:
//        scores (WMMA) -> LDS fp32 (16x4096 = 256KB, CDNA5 320KB LDS)
//        softmax in LDS -> nontemporal write of attention to d_out
//        O = Att x V^T (WMMA, K-loop 4096) -> out = x + O (nontemporal)
// ---------------------------------------------------------------------------

#define NB   4
#define CC   256
#define HW   4096
#define OCh  32

typedef __attribute__((ext_vector_type(16))) __bf16 v16bf;
typedef __attribute__((ext_vector_type(8)))  float  v8f;
typedef __attribute__((ext_vector_type(4)))  float  v4f;
typedef __attribute__((ext_vector_type(2)))  __bf16 v2bf;
typedef unsigned short u16;
typedef unsigned int   u32;

union BF16Frag { v16bf v; uint4 q[2]; };

__device__ inline u16 f2bf(float f) {
  u32 u = __float_as_uint(f);
  u += 0x7FFFu + ((u >> 16) & 1u);   // round-to-nearest-even
  return (u16)(u >> 16);
}

__device__ inline u32 pkbf(float lo, float hi) {
#if __has_builtin(__builtin_amdgcn_cvt_pk_bf16_f32)
  v2bf r = __builtin_amdgcn_cvt_pk_bf16_f32(lo, hi);
  return __builtin_bit_cast(u32, r);
#else
  return (u32)f2bf(lo) | ((u32)f2bf(hi) << 16);
#endif
}

// --------------------------- k0a: Wv -> bf16 --------------------------------
__global__ __launch_bounds__(256) void k0_wvbf(const float* __restrict__ wv,
                                               u16* __restrict__ wvbf) {
  int i = (blockIdx.x * 256 + threadIdx.x) * 2;   // 65536 elements / 2 per thread
  ((u32*)wvbf)[i >> 1] = pkbf(wv[i], wv[i + 1]);
}

// ----------------- k0b: x [n][c][p] -> xT bf16 [n][p][c] --------------------
__global__ __launch_bounds__(256) void k0_xt(const float* __restrict__ x,
                                             u16* __restrict__ xt) {
  int p = blockIdx.x * 256 + threadIdx.x;
  int n = blockIdx.y;
  const float* xp = x + (size_t)n * CC * HW + p;
  u32* o = (u32*)(xt + ((size_t)(n * HW + p)) * CC);
  for (int c = 0; c < CC; c += 2)
    o[c >> 1] = pkbf(xp[(size_t)c * HW], xp[(size_t)(c + 1) * HW]);
}

// -------------- k1: Q/K projection, fp32 accum -> bf16 [n][p][d] ------------
__global__ __launch_bounds__(256) void k1_qk(const float* __restrict__ x,
                                             const float* __restrict__ wq,
                                             const float* __restrict__ bq,
                                             const float* __restrict__ wk,
                                             const float* __restrict__ bk,
                                             u16* __restrict__ qbf,
                                             u16* __restrict__ kbf) {
  int p = blockIdx.x * 256 + threadIdx.x;
  int n = blockIdx.y;
  const float* xp = x + (size_t)n * CC * HW + p;
  float qa[OCh], ka[OCh];
#pragma unroll
  for (int d = 0; d < OCh; ++d) { qa[d] = bq[d]; ka[d] = bk[d]; }
  for (int c = 0; c < CC; ++c) {
    float xv = xp[(size_t)c * HW];
#pragma unroll
    for (int d = 0; d < OCh; ++d) {
      qa[d] = fmaf(wq[d * CC + c], xv, qa[d]);
      ka[d] = fmaf(wk[d * CC + c], xv, ka[d]);
    }
  }
  u32* qo = (u32*)(qbf + ((size_t)(n * HW + p)) * OCh);
  u32* ko = (u32*)(kbf + ((size_t)(n * HW + p)) * OCh);
#pragma unroll
  for (int d = 0; d < OCh; d += 2) {
    qo[d >> 1] = pkbf(qa[d], qa[d + 1]);
    ko[d >> 1] = pkbf(ka[d], ka[d + 1]);
  }
}

// ------------- k2: V = Wv * X + bv via WMMA -> bf16 [n][c][p] ---------------
__global__ __launch_bounds__(256) void k2_vproj(const u16* __restrict__ wvbf,
                                                const u16* __restrict__ xt,
                                                const float* __restrict__ bv,
                                                u16* __restrict__ vbf) {
  int lane = threadIdx.x & 31, wave = threadIdx.x >> 5;
  int hh = lane >> 4, lr = lane & 15;
  int gw  = blockIdx.x * 8 + wave;          // 16384 wave-tiles total
  int n   = gw >> 12;
  int rem = gw & 4095;
  int c0  = (rem >> 8) << 4;                // 16 c-tiles
  int p0  = (rem & 255) << 4;               // 256 p-tiles

  const uint4* wp = (const uint4*)(wvbf + (size_t)(c0 + lr) * CC);
  const uint4* bp = (const uint4*)(xt + ((size_t)(n * HW + p0 + lr)) * CC);
  v8f acc = {};
  for (int kb = 0; kb < CC; kb += 32) {
    int bi = kb >> 3;                       // uint4 index (8 bf16 each)
    BF16Frag a, b;
    a.q[0] = wp[bi + hh];     a.q[1] = wp[bi + 2 + hh];       // A: K {0..7,16..23}/{8..15,24..31}
    b.q[0] = bp[bi + hh];     b.q[1] = bp[bi + 2 + hh];       // B: K {0..7,16..23}/{8..15,24..31}
    acc = __builtin_amdgcn_wmma_f32_16x16x32_bf16(false, a.v, false, b.v,
                                                  (short)0, acc, false, false);
  }
#pragma unroll
  for (int v = 0; v < 8; ++v) {
    int c = c0 + v + 8 * hh;                // C: M = vgpr + 8*(lane/16)
    int p = p0 + lr;                        //    N = lane%16
    vbf[((size_t)(n * CC + c)) * HW + p] = f2bf(acc[v] + bv[c]);
  }
}

// ----------------------- k3: fused attention strip --------------------------
__global__ __launch_bounds__(256) void k3_attn(const float* __restrict__ x,
                                               const u16* __restrict__ qbf,
                                               const u16* __restrict__ kbf,
                                               const u16* __restrict__ vbf,
                                               float* __restrict__ out1,
                                               float* __restrict__ att) {
  extern __shared__ float smem[];
  float* sc     = smem;              // [16][4096] scores/probs (256 KB)
  float* part   = smem + 16 * HW;    // [256] reduction scratch
  float* rowmax = part + 256;        // [16]
  float* rowsum = rowmax + 16;       // [16]

  const int tid = threadIdx.x, lane = tid & 31, wave = tid >> 5;
  const int hh = lane >> 4, lr = lane & 15;
  const int n = blockIdx.y, i0 = blockIdx.x * 16;

  // ---- Phase A: scores S[16][4096] = Q_strip^T x K (one WMMA per 16x16 tile)
  BF16Frag a;                                   // Q strip A-fragment (shared)
  {
    const uint4* qp = (const uint4*)(qbf + ((size_t)(n * HW + i0 + lr)) * OCh);
    a.q[0] = qp[hh]; a.q[1] = qp[2 + hh];
  }
  for (int t = wave; t < 256; t += 8) {         // j-tiles round-robin per wave
    int j0 = t << 4;
    BF16Frag b;
    const uint4* kp = (const uint4*)(kbf + ((size_t)(n * HW + j0 + lr)) * OCh);
    b.q[0] = kp[hh]; b.q[1] = kp[2 + hh];
    v8f cc = {};
    cc = __builtin_amdgcn_wmma_f32_16x16x32_bf16(false, a.v, false, b.v,
                                                 (short)0, cc, false, false);
#pragma unroll
    for (int v = 0; v < 8; ++v)
      sc[(v + 8 * hh) * HW + j0 + lr] = cc[v];
  }
  __syncthreads();

  // ---- Phase B: softmax over 4096 keys per row (vectorized b128 LDS/global)
  // thread (r,s) owns j = k*64 + s*4 (+0..3): 16 lanes -> 256B contiguous
  const int r = tid >> 4, s = tid & 15;
  float* srow = sc + r * HW;
  float m = -3.402823e38f;
  for (int k = 0; k < 64; ++k) {
    v4f f = *(const v4f*)(srow + (k << 6) + (s << 2));
    m = fmaxf(m, fmaxf(fmaxf(f.x, f.y), fmaxf(f.z, f.w)));
  }
  part[tid] = m;
  __syncthreads();
  if (s == 0) {
    float mm = part[r << 4];
    for (int u = 1; u < 16; ++u) mm = fmaxf(mm, part[(r << 4) + u]);
    rowmax[r] = mm;
  }
  __syncthreads();
  const float M = rowmax[r];
  float sum = 0.f;
  for (int k = 0; k < 64; ++k) {               // exp once, keep in LDS
    v4f* p4 = (v4f*)(srow + (k << 6) + (s << 2));
    v4f f = *p4;
    f.x = __expf(f.x - M); f.y = __expf(f.y - M);
    f.z = __expf(f.z - M); f.w = __expf(f.w - M);
    *p4 = f;
    sum += (f.x + f.y) + (f.z + f.w);
  }
  part[tid] = sum;
  __syncthreads();
  if (s == 0) {
    float ss = 0.f;
    for (int u = 0; u < 16; ++u) ss += part[(r << 4) + u];
    rowsum[r] = ss;
  }
  __syncthreads();
  const float inv = 1.f / rowsum[r];
  float* arow = att + ((size_t)(n * HW + i0 + r)) * HW;
  for (int k = 0; k < 64; ++k) {               // normalize + streaming store
    v4f* p4 = (v4f*)(srow + (k << 6) + (s << 2));
    v4f f = *p4 * inv;
    *p4 = f;                                   // normalized probs stay in LDS
    __builtin_nontemporal_store(f, (v4f*)(arow + (k << 6) + (s << 2)));
  }
  __syncthreads();

  // ---- Phase C: O^T[16][256] = Att[16][4096] x V^T[4096][256]
  v8f acc0 = {}, acc1 = {};
  const int cA = wave << 4;                    // each wave owns 2 c-tiles
  const int cB = (wave + 8) << 4;
  const float* prow = sc + lr * HW;            // A rows: M = lane%16
  for (int kb = 0; kb < HW; kb += 32) {
    const float* p0f = prow + kb + 16 * hh;        // K {0..7}/{16..23}
    const float* p1f = prow + kb + 16 * hh + 8;    // K {8..15}/{24..31}
    v4f f0 = *(const v4f*)p0f;
    v4f f1 = *(const v4f*)(p0f + 4);
    v4f f2 = *(const v4f*)p1f;
    v4f f3 = *(const v4f*)(p1f + 4);
    BF16Frag pa;
    pa.q[0].x = pkbf(f0.x, f0.y); pa.q[0].y = pkbf(f0.z, f0.w);
    pa.q[0].z = pkbf(f1.x, f1.y); pa.q[0].w = pkbf(f1.z, f1.w);
    pa.q[1].x = pkbf(f2.x, f2.y); pa.q[1].y = pkbf(f2.z, f2.w);
    pa.q[1].z = pkbf(f3.x, f3.y); pa.q[1].w = pkbf(f3.z, f3.w);

    BF16Frag b0, b1;
    const uint4* v0p =
        (const uint4*)(vbf + ((size_t)(n * CC + cA + lr)) * HW + kb + hh * 16);
    b0.q[0] = v0p[0]; b0.q[1] = v0p[1];
    const uint4* v1p =
        (const uint4*)(vbf + ((size_t)(n * CC + cB + lr)) * HW + kb + hh * 16);
    b1.q[0] = v1p[0]; b1.q[1] = v1p[1];

    acc0 = __builtin_amdgcn_wmma_f32_16x16x32_bf16(false, pa.v, false, b0.v,
                                                   (short)0, acc0, false, false);
    acc1 = __builtin_amdgcn_wmma_f32_16x16x32_bf16(false, pa.v, false, b1.v,
                                                   (short)0, acc1, false, false);
  }

  // ---- Epilogue: out = x + O.  Lane's 8 acc elements are 8 consecutive i's:
  // i = i0 + 8*hh + v  -> two b128 loads of x, two nontemporal b128 stores.
  const size_t baseA = ((size_t)(n * CC + cA + lr)) * HW + i0 + 8 * hh;
  const size_t baseB = ((size_t)(n * CC + cB + lr)) * HW + i0 + 8 * hh;
  v4f xa0 = *(const v4f*)(x + baseA);
  v4f xa1 = *(const v4f*)(x + baseA + 4);
  v4f xb0 = *(const v4f*)(x + baseB);
  v4f xb1 = *(const v4f*)(x + baseB + 4);
  v4f oa0 = {acc0[0] + xa0.x, acc0[1] + xa0.y, acc0[2] + xa0.z, acc0[3] + xa0.w};
  v4f oa1 = {acc0[4] + xa1.x, acc0[5] + xa1.y, acc0[6] + xa1.z, acc0[7] + xa1.w};
  v4f ob0 = {acc1[0] + xb0.x, acc1[1] + xb0.y, acc1[2] + xb0.z, acc1[3] + xb0.w};
  v4f ob1 = {acc1[4] + xb1.x, acc1[5] + xb1.y, acc1[6] + xb1.z, acc1[7] + xb1.w};
  __builtin_nontemporal_store(oa0, (v4f*)(out1 + baseA));
  __builtin_nontemporal_store(oa1, (v4f*)(out1 + baseA + 4));
  __builtin_nontemporal_store(ob0, (v4f*)(out1 + baseB));
  __builtin_nontemporal_store(ob1, (v4f*)(out1 + baseB + 4));
}

// ---------------------------------------------------------------------------
extern "C" void kernel_launch(void* const* d_in, const int* in_sizes, int n_in,
                              void* d_out, int out_size, void* d_ws, size_t ws_size,
                              hipStream_t stream) {
  (void)in_sizes; (void)n_in; (void)out_size; (void)ws_size;
  const float* x  = (const float*)d_in[0];
  const float* Wq = (const float*)d_in[1];
  const float* bq = (const float*)d_in[2];
  const float* Wk = (const float*)d_in[3];
  const float* bk = (const float*)d_in[4];
  const float* Wv = (const float*)d_in[5];
  const float* bv = (const float*)d_in[6];

  float* out1 = (float*)d_out;                        // [4][256][4096]
  float* att  = out1 + (size_t)NB * CC * HW;          // [4][4096][4096]

  u16* ws   = (u16*)d_ws;
  u16* wvbf = ws;                                     //   65536 bf16 (128 KB)
  u16* xt   = wvbf + (size_t)CC * CC;                 // 4194304 bf16 (8 MB)
  u16* qbf  = xt + (size_t)NB * HW * CC;              //  524288 bf16 (1 MB)
  u16* kbf  = qbf + (size_t)NB * HW * OCh;            //  524288 bf16 (1 MB)
  u16* vbf  = kbf + (size_t)NB * HW * OCh;            // 4194304 bf16 (8 MB)

  k0_wvbf<<<128, 256, 0, stream>>>(Wv, wvbf);
  k0_xt<<<dim3(HW / 256, NB), 256, 0, stream>>>(x, xt);
  k1_qk<<<dim3(HW / 256, NB), 256, 0, stream>>>(x, Wq, bq, Wk, bk, qbf, kbf);
  k2_vproj<<<2048, 256, 0, stream>>>(wvbf, xt, bv, vbf);

  const size_t smem = (size_t)(16 * HW + 256 + 32) * sizeof(float); // ~257 KB < 320 KB
  k3_attn<<<dim3(HW / 16, NB), 256, smem, stream>>>(x, qbf, kbf, vbf, out1, att);
}
